// LMHAttention_37512244363843
// MI455X (gfx1250) — matrix operand order repaired and checked by reading, not compiled, not run on hardware
//
#include <hip/hip_runtime.h>
#include <hip/hip_bf16.h>
#include <stdint.h>

#define DEVINL __device__ __forceinline__

typedef _Float16 half8  __attribute__((ext_vector_type(8)));
typedef _Float16 half16 __attribute__((ext_vector_type(16)));
typedef float    f32x8  __attribute__((ext_vector_type(8)));
typedef unsigned int u32x4 __attribute__((ext_vector_type(4)));
typedef int          i32x4 __attribute__((ext_vector_type(4)));
typedef int          i32x8 __attribute__((ext_vector_type(8)));

constexpr int NB = 32, SEQ = 512, NH = 16, HD = 64, DMODEL = 1024;

#if __has_builtin(__builtin_amdgcn_tensor_load_to_lds) && \
    __has_builtin(__builtin_amdgcn_s_wait_tensorcnt)
#define USE_TDM 1
#else
#define USE_TDM 0
#endif

// ---------------- WMMA core (CDNA5, wave32) ----------------
DEVINL f32x8 wmma_f16(half16 a, half16 b, f32x8 c) {
  // 8 args: (neg_a, A, neg_b, B, c_mod, C, reuse_a, reuse_b)
  return __builtin_amdgcn_wmma_f32_16x16x32_f16(false, a, false, b, (short)0, c, false, false);
}

// A fragment 16x32 f16 from row-major A[M,K] (ISA 7.12.2 16-bit A layout):
// lane<16: M=lane, K=k0+{0..7,16..23}; lane>=16: M=lane-16, K=k0+{8..15,24..31}
DEVINL half16 load_fragA_h(const _Float16* A, int m0, int ld, int k0) {
  const int lane = threadIdx.x & 31;
  const int row  = m0 + (lane & 15);
  const int kb   = (lane & 16) ? 8 : 0;
  const _Float16* p = A + (size_t)row * ld + (size_t)(k0 + kb);
  half8 lo = *(const half8*)(p);
  half8 hi = *(const half8*)(p + 16);
  return __builtin_shufflevector(lo, hi, 0,1,2,3,4,5,6,7,8,9,10,11,12,13,14,15);
}

// Same, but source is f32 (attention probabilities) converted in-register
DEVINL half16 load_fragA_f32(const float* A, int m0, int ld, int k0) {
  const int lane = threadIdx.x & 31;
  const int row  = m0 + (lane & 15);
  const int kb   = (lane & 16) ? 8 : 0;
  const float* p = A + (size_t)row * ld + (size_t)(k0 + kb);
  f32x8 lo = *(const f32x8*)(p);
  f32x8 hi = *(const f32x8*)(p + 16);
  half16 r;
#pragma unroll
  for (int i = 0; i < 8; ++i) { r[i] = (_Float16)lo[i]; r[i + 8] = (_Float16)hi[i]; }
  return r;
}

// B fragment 32x16 f16 sourced from Bt row-major [N,K] (same K order as A:
// lane<16: N=lane, K=k0+{0..7,16..23}; lane>=16: K=k0+{8..15,24..31})
DEVINL half16 load_fragB_h(const _Float16* Bt, int n0, int ld, int k0) {
  const int lane = threadIdx.x & 31;
  const int n    = n0 + (lane & 15);
  const int kb   = (lane & 16) ? 8 : 0;
  const _Float16* p = Bt + (size_t)n * ld + (size_t)(k0 + kb);
  half8 lo = *(const half8*)(p);
  half8 hi = *(const half8*)(p + 16);
  return __builtin_shufflevector(lo, hi, 0,1,2,3,4,5,6,7,8,9,10,11,12,13,14,15);
}

#if USE_TDM
// Issue a TDM 2D tile load (f16 elements) into LDS.
// D# per ISA ch.8: group0 = {count/lds_addr/global_addr/type}, group1 = dims.
// This toolchain exposes the 6-arg builtin: (g0, g1, g2, g3, g4, cpol);
// groups 2+ are zero (2D tile, tile_dim2=0, no iteration/gather).
DEVINL void tdm_load_tile_f16(uint32_t lds_byte_addr, const _Float16* gsrc,
                              uint32_t tile_w, uint32_t tile_h, uint32_t row_stride) {
  const uint64_t ga = (uint64_t)(uintptr_t)gsrc;
  u32x4 g0;
  g0[0] = 1u;                                             // count=1 (valid), user mode
  g0[1] = lds_byte_addr;                                  // lds_addr [63:32]
  g0[2] = (uint32_t)(ga & 0xffffffffu);                   // global_addr [95:64]
  g0[3] = (uint32_t)((ga >> 32) & 0x01ffffffu)            // global_addr [120:96]
        | (2u << 30);                                     // type=2 ("image")
  i32x8 g1;
  g1[0] = (int)(1u << 16);                                // data_size=1 (2 bytes)
  g1[1] = (int)((tile_w & 0xffffu) << 16);                // tensor_dim0[15:0]
  g1[2] = (int)(((tile_w >> 16) & 0xffffu) |              // tensor_dim0[31:16]
                ((tile_h & 0xffffu) << 16));              // tensor_dim1[15:0]
  g1[3] = (int)(((tile_h >> 16) & 0xffffu) |              // tensor_dim1[31:16]
                ((tile_w & 0xffffu) << 16));              // tile_dim0
  g1[4] = (int)(tile_h & 0xffffu);                        // tile_dim1 (tile_dim2=0)
  g1[5] = (int)row_stride;                                // tensor_dim0_stride[31:0]
  g1[6] = 0;                                              // stride[47:32], dim1_stride lo
  g1[7] = 0;
  const i32x4 gz4 = {0, 0, 0, 0};
  const i32x8 gz8 = {0, 0, 0, 0, 0, 0, 0, 0};
  __builtin_amdgcn_tensor_load_to_lds(g0, g1, gz4, gz4, gz8, 0);
}
#endif

// ---------------- f32 -> f16 conversion ----------------
__global__ void cvt_f32_f16_kernel(const float* __restrict__ src,
                                   _Float16* __restrict__ dst, int n) {
  int i = blockIdx.x * blockDim.x + threadIdx.x;
  if (i < n) dst[i] = (_Float16)src[i];
}

// ---------------- Kernel 1: per-head QKV projections ----------------
// grid.x = NB*NH*3, block = 256 (8 waves). GEMM [512,64] x Bt[64,64].
// Q stored row-major [S,HD]; K,V stored transposed [HD,S] for the seq-dim GEMMs.
__global__ __launch_bounds__(256)
void qkv_proj_kernel(const _Float16* __restrict__ xq, const _Float16* __restrict__ xk,
                     const _Float16* __restrict__ xv,
                     const _Float16* __restrict__ Wqf, const _Float16* __restrict__ Wkf,
                     const _Float16* __restrict__ Wvf,
                     const float* __restrict__ bq, const float* __restrict__ bk,
                     const float* __restrict__ bv,
                     _Float16* __restrict__ Q, _Float16* __restrict__ Kt,
                     _Float16* __restrict__ Vt) {
  int id = blockIdx.x;
  const int which = id % 3; id /= 3;
  const int h = id % NH;
  const int b = id / NH;
  const _Float16* X; const _Float16* W; const float* bias;
  if (which == 0)      { X = xq; W = Wqf; bias = bq; }
  else if (which == 1) { X = xk; W = Wkf; bias = bk; }
  else                 { X = xv; W = Wvf; bias = bv; }
  X += (size_t)b * SEQ * HD;        // A: [512,64]
  W += (size_t)h * HD * HD;         // Bt: [64,64] (torch Linear weight [out,in])
  bias += (size_t)h * HD;
  const size_t bh = (size_t)b * NH + h;
  const int wave = threadIdx.x >> 5;
  const int lane = threadIdx.x & 31;

  for (int t = wave; t < 32 * 4; t += 8) {          // 32 M-tiles x 4 N-tiles
    const int m0 = (t >> 2) * 16, n0 = (t & 3) * 16;
    f32x8 acc = {};
#pragma unroll
    for (int k0 = 0; k0 < HD; k0 += 32)
      acc = wmma_f16(load_fragA_h(X, m0, HD, k0), load_fragB_h(W, n0, HD, k0), acc);

    const int n  = n0 + (lane & 15);
    const int rb = m0 + ((lane & 16) ? 8 : 0);
    const float bb = bias[n];
    if (which == 0) {
      _Float16* D = Q + bh * (size_t)SEQ * HD;      // row-major [S,HD]
#pragma unroll
      for (int r = 0; r < 8; ++r)
        D[(size_t)(rb + r) * HD + n] = (_Float16)(acc[r] + bb);
    } else {
      _Float16* D = (which == 1 ? Kt : Vt) + bh * (size_t)HD * SEQ;  // [HD,S]
      half8 v;
#pragma unroll
      for (int r = 0; r < 8; ++r) v[r] = (_Float16)(acc[r] + bb);
      *(half8*)(D + (size_t)n * SEQ + rb) = v;      // contiguous along S
    }
  }
}

// ---------------- Kernel 2: seq-dim projections (Linformer E/F) ----------------
// which=0: KEt[t,d] = sum_s Ew[t,s]*Kt[d,s] + E_b[t]   (A=Ew [512,512], Bt=Kt [64,512])
// which=1: VFt[d,t] = sum_s Vt[d,s]*Fw[t,s] + F_b[t]   (A=Vt [64,512],  Bt=Fw [512,512])
__global__ __launch_bounds__(256)
void seqproj_kernel(const _Float16* __restrict__ Ewf, const _Float16* __restrict__ Fwf,
                    const float* __restrict__ E_b, const float* __restrict__ F_b,
                    const _Float16* __restrict__ Kt, const _Float16* __restrict__ Vt,
                    _Float16* __restrict__ KEt, _Float16* __restrict__ VFt) {
  int id = blockIdx.x;
  const int which = id & 1; id >>= 1;
  const size_t bh = (size_t)id;
  const int wave = threadIdx.x >> 5;
  const int lane = threadIdx.x & 31;

  if (which == 0) {
    const _Float16* A  = Ewf;
    const _Float16* Bt = Kt + bh * (size_t)HD * SEQ;
    _Float16* D = KEt + bh * (size_t)SEQ * HD;      // row-major [T,HD]
    for (int t = wave; t < 32 * 4; t += 8) {
      const int m0 = (t >> 2) * 16, n0 = (t & 3) * 16;
      f32x8 acc = {};
#pragma unroll 4
      for (int k0 = 0; k0 < SEQ; k0 += 32)
        acc = wmma_f16(load_fragA_h(A, m0, SEQ, k0), load_fragB_h(Bt, n0, SEQ, k0), acc);
      const int n  = n0 + (lane & 15);
      const int rb = m0 + ((lane & 16) ? 8 : 0);
#pragma unroll
      for (int r = 0; r < 8; ++r)                   // bias along M (=t)
        D[(size_t)(rb + r) * HD + n] = (_Float16)(acc[r] + E_b[rb + r]);
    }
  } else {
    const _Float16* A  = Vt + bh * (size_t)HD * SEQ;
    const _Float16* Bt = Fwf;
    _Float16* D = VFt + bh * (size_t)HD * SEQ;      // row-major [HD,T]
    for (int t = wave; t < 4 * 32; t += 8) {
      const int m0 = (t >> 5) * 16, n0 = (t & 31) * 16;
      f32x8 acc = {};
#pragma unroll 4
      for (int k0 = 0; k0 < SEQ; k0 += 32)
        acc = wmma_f16(load_fragA_h(A, m0, SEQ, k0), load_fragB_h(Bt, n0, SEQ, k0), acc);
      const int n  = n0 + (lane & 15);
      const int rb = m0 + ((lane & 16) ? 8 : 0);
      const float bb = F_b[n];                      // bias along N (=t)
#pragma unroll
      for (int r = 0; r < 8; ++r)
        D[(size_t)(rb + r) * SEQ + n] = (_Float16)(acc[r] + bb);
    }
  }
}

// ---------------- Kernel 3: scores = Q*KEt^T, softmax over T ----------------
// grid.x = NB*NH*(SEQ/16), block 256 (8 waves). Scores staged in 32KB LDS.
__global__ __launch_bounds__(256)
void attn_softmax_kernel(const _Float16* __restrict__ Q, const _Float16* __restrict__ KEt,
                         float* __restrict__ attn_out) {
  __shared__ float sc[16][SEQ];
  int id = blockIdx.x;
  const int mt = id & 31; id >>= 5;
  const size_t bh = (size_t)id;
  const _Float16* A  = Q   + bh * (size_t)SEQ * HD;   // [512,64]
  const _Float16* Bt = KEt + bh * (size_t)SEQ * HD;   // [512,64] rows = t
  const int wave = threadIdx.x >> 5;
  const int lane = threadIdx.x & 31;
  const int m0 = mt * 16;

  const half16 a0 = load_fragA_h(A, m0, HD, 0);       // K = HD = 64: two k-steps
  const half16 a1 = load_fragA_h(A, m0, HD, 32);
  for (int nt = wave; nt < 32; nt += 8) {
    const int n0 = nt * 16;
    f32x8 acc = {};
    acc = wmma_f16(a0, load_fragB_h(Bt, n0, HD, 0), acc);
    acc = wmma_f16(a1, load_fragB_h(Bt, n0, HD, 32), acc);
    const int n  = n0 + (lane & 15);
    const int rb = (lane & 16) ? 8 : 0;
#pragma unroll
    for (int r = 0; r < 8; ++r) sc[rb + r][n] = acc[r] * 0.125f;  // 1/sqrt(64)
  }
  __syncthreads();

  // softmax: wave w owns rows {w, w+8}; 32 lanes cover 512 cols, wave32 shuffles
  for (int row = wave; row < 16; row += 8) {
    float v[16];
    float mx = -3.402823e38f;
#pragma unroll
    for (int i = 0; i < 16; ++i) { v[i] = sc[row][lane + 32 * i]; mx = fmaxf(mx, v[i]); }
#pragma unroll
    for (int off = 16; off >= 1; off >>= 1) mx = fmaxf(mx, __shfl_xor(mx, off, 32));
    float sum = 0.f;
#pragma unroll
    for (int i = 0; i < 16; ++i) { v[i] = __expf(v[i] - mx); sum += v[i]; }
#pragma unroll
    for (int off = 16; off >= 1; off >>= 1) sum += __shfl_xor(sum, off, 32);
    const float inv = 1.0f / sum;
    float* out = attn_out + (bh * SEQ + (size_t)(m0 + row)) * SEQ;
#pragma unroll
    for (int i = 0; i < 16; ++i) out[lane + 32 * i] = v[i] * inv;
  }
}

// ---------------- Kernel 4: head_out = attn * VFt^T -> mha ----------------
// grid.x = NB*NH*(SEQ/16), block 128 (4 waves, one N-tile each).
__global__ __launch_bounds__(128)
void headout_kernel(const float* __restrict__ attn, const _Float16* __restrict__ VFt,
                    _Float16* __restrict__ mha) {
  int id = blockIdx.x;
  const int mt = id & 31; id >>= 5;
  const int h = id % NH, b = id / NH;
  const size_t bh = (size_t)b * NH + h;
  const float* A     = attn + bh * (size_t)SEQ * SEQ;   // [512,512] f32
  const _Float16* Bt = VFt  + bh * (size_t)HD * SEQ;    // [64,512]
  const int wave = threadIdx.x >> 5;
  const int lane = threadIdx.x & 31;
  const int m0 = mt * 16, n0 = wave * 16;

  f32x8 acc = {};
#pragma unroll 4
  for (int k0 = 0; k0 < SEQ; k0 += 32)
    acc = wmma_f16(load_fragA_f32(A, m0, SEQ, k0), load_fragB_h(Bt, n0, SEQ, k0), acc);

  const int n  = n0 + (lane & 15);
  const int rb = m0 + ((lane & 16) ? 8 : 0);
  const int e  = h * HD + n;                            // concat heads
#pragma unroll
  for (int r = 0; r < 8; ++r)
    mha[((size_t)b * SEQ + rb + r) * DMODEL + e] = (_Float16)acc[r];
}

// ---------------- Kernel 5: out = mha * out_w^T + out_b ----------------
// [16384,1024] x [1024,1024]; block 256 = 8 waves, each a 16x32 strip of a
// 16x256 block-tile. A k-slice (16x32 f16) is staged in LDS by the Tensor Data
// Mover, double-buffered: wave0 issues tensor_load_to_lds for step i+1 while
// all waves compute step i (TENSORcnt + split barriers).
__global__ __launch_bounds__(256)
void outproj_kernel(const _Float16* __restrict__ mha, const _Float16* __restrict__ Wof,
                    const float* __restrict__ out_b, float* __restrict__ out) {
  __shared__ __align__(16) _Float16 As[2][16 * 32];
  int id = blockIdx.x;
  const int nblk = id & 3;                 // 4 N-blocks of 256 columns
  const int m0 = (id >> 2) * 16;
  const int wave = threadIdx.x >> 5;
  const int lane = threadIdx.x & 31;
  const int n0 = nblk * 256 + wave * 32;
  const _Float16* Arow = mha + (size_t)m0 * DMODEL;

  f32x8 acc0 = {}, acc1 = {};
#if USE_TDM
  if (wave == 0)
    tdm_load_tile_f16((uint32_t)(size_t)&As[0][0], Arow, 32, 16, DMODEL);
  for (int ks = 0; ks < 32; ++ks) {
    if (wave == 0) __builtin_amdgcn_s_wait_tensorcnt(0);   // current buffer ready
    __syncthreads();                                       // publish LDS to all waves
    if (wave == 0 && ks + 1 < 32)                          // prefetch next k-slice
      tdm_load_tile_f16((uint32_t)(size_t)&As[(ks + 1) & 1][0],
                        Arow + (size_t)(ks + 1) * 32, 32, 16, DMODEL);
    const int k0 = ks * 32;
    __builtin_prefetch(Wof + (size_t)n0 * DMODEL + k0 + 128, 0, 1);
    const half16 a = load_fragA_h(&As[ks & 1][0], 0, 32, 0);
    acc0 = wmma_f16(a, load_fragB_h(Wof, n0, DMODEL, k0), acc0);
    acc1 = wmma_f16(a, load_fragB_h(Wof, n0 + 16, DMODEL, k0), acc1);
    __syncthreads();                       // reads done before buffer reuse
  }
#else
  for (int ks = 0; ks < 32; ++ks) {
    const int k0 = ks * 32;
    __syncthreads();
    {   // 256 threads load 16x32 halfs (one u32 each)
      const int rr = threadIdx.x >> 4;
      const int cc = (threadIdx.x & 15) * 2;
      *(uint32_t*)(&As[0][rr * 32 + cc]) =
          *(const uint32_t*)(Arow + (size_t)rr * DMODEL + k0 + cc);
    }
    __syncthreads();
    const half16 a = load_fragA_h(&As[0][0], 0, 32, 0);
    acc0 = wmma_f16(a, load_fragB_h(Wof, n0, DMODEL, k0), acc0);
    acc1 = wmma_f16(a, load_fragB_h(Wof, n0 + 16, DMODEL, k0), acc1);
  }
#endif
  const int rb = m0 + ((lane & 16) ? 8 : 0);
  {
    const int n = n0 + (lane & 15);
    const float bb = out_b[n];
#pragma unroll
    for (int r = 0; r < 8; ++r) out[(size_t)(rb + r) * DMODEL + n] = acc0[r] + bb;
  }
  {
    const int n = n0 + 16 + (lane & 15);
    const float bb = out_b[n];
#pragma unroll
    for (int r = 0; r < 8; ++r) out[(size_t)(rb + r) * DMODEL + n] = acc1[r] + bb;
  }
}

// ---------------- host launch ----------------
extern "C" void kernel_launch(void* const* d_in, const int* in_sizes, int n_in,
                              void* d_out, int out_size, void* d_ws, size_t ws_size,
                              hipStream_t stream) {
  const float* query = (const float*)d_in[0];
  const float* key   = (const float*)d_in[1];
  const float* value = (const float*)d_in[2];
  const float* Wq    = (const float*)d_in[3];
  const float* bq    = (const float*)d_in[4];
  const float* Wk    = (const float*)d_in[5];
  const float* bk    = (const float*)d_in[6];
  const float* Wv    = (const float*)d_in[7];
  const float* bv    = (const float*)d_in[8];
  const float* E_w   = (const float*)d_in[9];
  const float* E_b   = (const float*)d_in[10];
  const float* F_w   = (const float*)d_in[11];
  const float* F_b   = (const float*)d_in[12];
  const float* out_w = (const float*)d_in[13];
  const float* out_b = (const float*)d_in[14];

  float* mha_out  = (float*)d_out;                                  // [B,S,DM]
  float* attn_out = (float*)d_out + (size_t)NB * SEQ * DMODEL;      // [B,H,S,T]

  // workspace layout (all f16)
  _Float16* ws = (_Float16*)d_ws;
  size_t off = 0;
  auto take = [&](size_t n) { _Float16* p = ws + off; off += n; return p; };
  const size_t N_X = (size_t)NB * SEQ * HD;
  const size_t N_W = (size_t)NH * HD * HD;
  const size_t N_E = (size_t)SEQ * SEQ;
  const size_t N_O = (size_t)DMODEL * DMODEL;
  const size_t N_Q = (size_t)NB * NH * SEQ * HD;
  _Float16* qf   = take(N_X);
  _Float16* kf   = take(N_X);
  _Float16* vf   = take(N_X);
  _Float16* Wqf  = take(N_W);
  _Float16* Wkf  = take(N_W);
  _Float16* Wvf  = take(N_W);
  _Float16* Ewf  = take(N_E);
  _Float16* Fwf  = take(N_E);
  _Float16* Wof  = take(N_O);
  _Float16* Qb   = take(N_Q);   // [B,H,S,HD]
  _Float16* Ktb  = take(N_Q);   // [B,H,HD,S]
  _Float16* Vtb  = take(N_Q);   // [B,H,HD,S]
  _Float16* KEt  = take(N_Q);   // [B,H,T,HD]
  _Float16* VFt  = take(N_Q);   // [B,H,HD,T]
  _Float16* mhah = take((size_t)NB * SEQ * DMODEL);
  (void)ws_size; (void)n_in; (void)in_sizes; (void)out_size;

  auto cvt = [&](const float* s, _Float16* d, size_t n) {
    cvt_f32_f16_kernel<<<(unsigned)((n + 255) / 256), 256, 0, stream>>>(s, d, (int)n);
  };
  cvt(query, qf, N_X); cvt(key, kf, N_X); cvt(value, vf, N_X);
  cvt(Wq, Wqf, N_W);   cvt(Wk, Wkf, N_W); cvt(Wv, Wvf, N_W);
  cvt(E_w, Ewf, N_E);  cvt(F_w, Fwf, N_E); cvt(out_w, Wof, N_O);

  qkv_proj_kernel<<<NB * NH * 3, 256, 0, stream>>>(qf, kf, vf, Wqf, Wkf, Wvf,
                                                   bq, bk, bv, Qb, Ktb, Vtb);
  seqproj_kernel<<<NB * NH * 2, 256, 0, stream>>>(Ewf, Fwf, E_b, F_b, Ktb, Vtb, KEt, VFt);
  attn_softmax_kernel<<<NB * NH * (SEQ / 16), 256, 0, stream>>>(Qb, KEt, attn_out);
  headout_kernel<<<NB * NH * (SEQ / 16), 128, 0, stream>>>(attn_out, VFt, mhah);
  outproj_kernel<<<(NB * SEQ / 16) * (DMODEL / 256), 256, 0, stream>>>(mhah, Wof, out_b, mha_out);
}
